// HardwareOptimizedSpikeProcessor_56470230008110
// MI455X (gfx1250) — 8-bit matrix operand order repaired and checked by reading, compiled, not run on hardware
//
#include <hip/hip_runtime.h>
#include <stdint.h>

#define B_SZ   64
#define CIN    2048
#define COUT   2048
#define T_STEPS 128
#define KTILES (CIN / 64)

typedef __attribute__((ext_vector_type(8))) int v8i;

__global__ __launch_bounds__(256) void pack_weights_kernel(
    const float* __restrict__ w, const float* __restrict__ m,
    int8_t* __restrict__ w8) {
  const int i = (blockIdx.x * 256 + threadIdx.x) * 4;
  float4 wv = *(const float4*)(w + i);
  float4 mv = *(const float4*)(m + i);
  int b0 = (int)(int8_t)(wv.x * mv.x) & 0xff;
  int b1 = (int)(int8_t)(wv.y * mv.y) & 0xff;
  int b2 = (int)(int8_t)(wv.z * mv.z) & 0xff;
  int b3 = (int)(int8_t)(wv.w * mv.w) & 0xff;
  *(int*)(w8 + i) = b0 | (b1 << 8) | (b2 << 16) | (b3 << 24);
}

__global__ __launch_bounds__(256) void pack_spikes_kernel(
    const float* __restrict__ sp, int8_t* __restrict__ s8) {
  __shared__ int8_t tile[128 * 132];
  const int b  = blockIdx.y;
  const int c0 = blockIdx.x * 128;
  const float* src = sp + ((size_t)b * CIN + c0) * T_STEPS;

  for (int i = threadIdx.x; i < 128 * 128 / 4; i += 256) {
    const int c  = i / 32;
    const int t4 = (i % 32) * 4;
    float4 v = *(const float4*)(src + (size_t)c * T_STEPS + t4);
    int8_t* d = &tile[c * 132 + t4];
    d[0] = (int8_t)v.x; d[1] = (int8_t)v.y;
    d[2] = (int8_t)v.z; d[3] = (int8_t)v.w;
  }
  __syncthreads();
  for (int i = threadIdx.x; i < 128 * 128; i += 256) {
    const int t = i / 128;
    const int c = i % 128;
    s8[(size_t)t * (B_SZ * CIN) + (size_t)b * CIN + c0 + c] = tile[c * 132 + t];
  }
}

__device__ __forceinline__ v8i load_a_tile(const int8_t* p) {
  int2 a01 = *(const int2*)(p);
  int2 a23 = *(const int2*)(p + 16);
  int2 a45 = *(const int2*)(p + 32);
  int2 a67 = *(const int2*)(p + 48);
  v8i a;
  a[0] = a01.x; a[1] = a01.y; a[2] = a23.x; a[3] = a23.y;
  a[4] = a45.x; a[5] = a45.y; a[6] = a67.x; a[7] = a67.y;
  return a;
}

__device__ __forceinline__ v8i load_b_tile(const int8_t* p) {
  int2 b01 = *(const int2*)(p);
  int2 b23 = *(const int2*)(p + 16);
  int2 b45 = *(const int2*)(p + 32);
  int2 b67 = *(const int2*)(p + 48);
  v8i b;
  b[0] = b01.x; b[1] = b01.y; b[2] = b23.x; b[3] = b23.y;
  b[4] = b45.x; b[5] = b45.y; b[6] = b67.x; b[7] = b67.y;
  return b;
}

__global__ __launch_bounds__(128, 1) void snn_wmma_kernel(
    const int8_t* __restrict__ w8, const int8_t* __restrict__ s8,
    const int* __restrict__ scale_exp, const int* __restrict__ thr_exp,
    float* __restrict__ out) {
  __shared__ int8_t lw[16 * CIN];

  const int cout0 = blockIdx.x * 16;
  const int tid   = threadIdx.x;
  const int lane  = tid & 31;
  const int wave  = tid >> 5;
  const int b0    = wave * 16;
  const int n     = lane & 15;
  const int half  = lane >> 4;

  {
    const int4* src = (const int4*)(w8 + (size_t)cout0 * CIN);
    int4* dst = (int4*)lw;
    for (int i = tid; i < 16 * CIN / 16; i += 128) dst[i] = src[i];
  }
  __syncthreads();

  const float scale = exp2f((float)scale_exp[cout0 + n]);
  const float thr   = exp2f((float)thr_exp[0]);

  const int8_t* aptr = s8 + (size_t)(b0 + n) * CIN + half * 8;
  const int8_t* bptr = lw + n * CIN + half * 8;

  float accf[8];
#pragma unroll
  for (int r = 0; r < 8; ++r) accf[r] = 0.0f;

  for (int wchunk = 0; wchunk < 4; ++wchunk) {
    unsigned bw[8];
#pragma unroll
    for (int r = 0; r < 8; ++r) bw[r] = 0u;

    for (int t2 = 0; t2 < 32; ++t2) {
      const int t = wchunk * 32 + t2;
      const int8_t* at = aptr + (size_t)t * (B_SZ * CIN);

      v8i ci0 = {0, 0, 0, 0, 0, 0, 0, 0};
      v8i ci1 = {0, 0, 0, 0, 0, 0, 0, 0};
#pragma unroll 4
      for (int kt = 0; kt < KTILES; kt += 2) {
        v8i a0 = load_a_tile(at + kt * 64);
        v8i b0v = load_b_tile(bptr + kt * 64);
        ci0 = __builtin_amdgcn_wmma_i32_16x16x64_iu8(
             false, a0,  true, b0v, ci0, false, false);
        v8i a1 = load_a_tile(at + (kt + 1) * 64);
        v8i b1v = load_b_tile(bptr + (kt + 1) * 64);
        ci1 = __builtin_amdgcn_wmma_i32_16x16x64_iu8(
             false, a1,  true, b1v, ci1, false, false);
      }

#pragma unroll
      for (int r = 0; r < 8; ++r) {
        const float v = accf[r] + (float)(ci0[r] + ci1[r]) * scale;
        const unsigned fired = (v >= thr) ? 1u : 0u;
        bw[r] |= fired << t2;
        accf[r] = fired ? 0.0f : v;
      }
    }

#pragma unroll
    for (int r = 0; r < 8; ++r) {
      const int brow = b0 + r + half * 8;
      float* op = out + ((size_t)brow * COUT + (cout0 + n)) * T_STEPS
                      + wchunk * 32;
      const unsigned bv = bw[r];
#pragma unroll
      for (int j = 0; j < 32; j += 4) {
        float4 o;
        o.x = (bv >> (j + 0)) & 1u ? 1.0f : 0.0f;
        o.y = (bv >> (j + 1)) & 1u ? 1.0f : 0.0f;
        o.z = (bv >> (j + 2)) & 1u ? 1.0f : 0.0f;
        o.w = (bv >> (j + 3)) & 1u ? 1.0f : 0.0f;
        *(float4*)(op + j) = o;
      }
    }
  }
}

extern "C" void kernel_launch(void* const* d_in, const int* in_sizes, int n_in,
                              void* d_out, int out_size, void* d_ws,
                              size_t ws_size, hipStream_t stream) {
  const float* spikes   = (const float*)d_in[0];
  const float* weights  = (const float*)d_in[1];
  const float* mask     = (const float*)d_in[2];
  const int*   scale_e  = (const int*)d_in[3];
  const int*   thr_e    = (const int*)d_in[4];
  float* out = (float*)d_out;

  int8_t* w8 = (int8_t*)d_ws;
  int8_t* s8 = (int8_t*)d_ws + (size_t)COUT * CIN;

  pack_weights_kernel<<<(COUT * CIN) / (256 * 4), 256, 0, stream>>>(
      weights, mask, w8);
  pack_spikes_kernel<<<dim3(CIN / 128, B_SZ), 256, 0, stream>>>(spikes, s8);
  snn_wmma_kernel<<<COUT / 16, 128, 0, stream>>>(w8, s8, scale_e, thr_e, out);
}
